// SEGNNLayer_11708080849361
// MI455X (gfx1250) — matrix operand order repaired and checked by reading, not compiled, not run on hardware
//
#include <hip/hip_runtime.h>

typedef __attribute__((ext_vector_type(16))) _Float16 v16h;
typedef __attribute__((ext_vector_type(8)))  _Float16 v8h;
typedef __attribute__((ext_vector_type(4)))  _Float16 v4h;
typedef __attribute__((ext_vector_type(8)))  float    v8f;

#define EW 2   // waves (16-row tiles) per block

// packed-weight offsets in d_ws, in halves (each frag = 512 halves = 32 lanes x 16)
#define OFF_M1S   0        // [xs(160) ; dotv(128)] -> 128 : NT=8 KT=9
#define OFF_M1SV  36864    // W1sv 160 -> 64        : NT=4 KT=5
#define OFF_M1VV  47104    // W1vv 128 -> 64        : NT=4 KT=4
#define OFF_M2S   55296    // [ms(64) ; dot2(64)] -> 128 : NT=8 KT=4
#define OFF_M2SV  71680    // W2sv 64 -> 64         : NT=4 KT=2
#define OFF_M2VV  75776    // W2vv 64 -> 64         : NT=4 KT=2
#define OFF_U0S   79872    // [xs(128); dot(128)] -> 128 : NT=8 KT=8
#define OFF_U0SV  112640   // u0sv 128 -> 64        : NT=4 KT=4
#define OFF_U0VV  120832   // u0vv 128 -> 64        : NT=4 KT=4
#define OFF_U1S   129024   // [hs(64); dot3(64)] -> 64 : NT=4 KT=4
#define OFF_U1SV  137216   // u1sv 64 -> 64         : NT=4 KT=2
#define OFF_U1VV  141312   // u1vv 64 -> 64         : NT=4 KT=2

#define INV_M1 0.06225728064f   // 1/sqrt(130+128)
#define INV_M2 0.08838834765f   // 1/sqrt(64+64)
#define INV_U0 0.0625f          // 1/sqrt(128+128)
#define INV_U1 0.08838834765f   // 1/sqrt(64+64)
#define RSQRT3 0.57735026919f

#define WMMA(a, b, c) __builtin_amdgcn_wmma_f32_16x16x32_f16(false, (a), false, (b), (short)0, (c), false, false)

__device__ __forceinline__ void wave_sync() {
  __builtin_amdgcn_fence(__ATOMIC_ACQ_REL, "workgroup");
  __builtin_amdgcn_wave_barrier();
}

__device__ __forceinline__ float sigmoidf_(float x) { return 1.f / (1.f + __expf(-x)); }

// A fragment (16x32 f16, row-major source in LDS). CDNA5 layout: lanes 0-15 row M=lane,
// elements 0..7 = K k0..k0+7, 8..15 = K k0+16..k0+23 ; lanes 16-31 same row, K base +8.
__device__ __forceinline__ v16h frag_a(const _Float16* p, int ld, int k0, int lane) {
  const int row = lane & 15;
  const int kb  = k0 + ((lane & 16) ? 8 : 0);
  const _Float16* q = p + row * ld + kb;
  v8h lo = *(const v8h*)(q);
  v8h hi = *(const v8h*)(q + 16);
  v16h a;
#pragma unroll
  for (int i = 0; i < 8; ++i) { a[i] = lo[i]; a[i + 8] = hi[i]; }
  return a;
}

// B fragment pre-packed: frag*512 + lane*16 contiguous halves (32B per lane).
__device__ __forceinline__ v16h frag_b(const _Float16* __restrict__ pkb, int frag, int lane) {
  return *(const v16h*)(pkb + (size_t)frag * 512 + (size_t)lane * 16);
}

// ---------------- weight pack: f32 [K,N] (optionally two stacked along K) -> B frags ----------
__global__ __launch_bounds__(512) void pack_kernel(
    const float* __restrict__ W0, int K0, int Kp0,
    const float* __restrict__ W1, int K1,
    int N, int KT, _Float16* __restrict__ dst)
{
  const int f    = blockIdx.x;
  const int kt   = f % KT;
  const int nt   = f / KT;
  const int lane = threadIdx.x >> 4;
  const int e    = threadIdx.x & 15;
  const int n    = nt * 16 + (lane & 15);
  const int kb   = (lane & 16) ? 8 : 0;    // B layout: lanes 0-15 K 0..7,16..23, lanes 16-31 K 8..15,24..31
  const int k    = kt * 32 + kb + (e & 7) + ((e & 8) ? 16 : 0);
  float v = 0.f;
  if (k < Kp0) { if (k < K0) v = W0[(size_t)k * N + n]; }
  else         { int k2 = k - Kp0; if (k2 < K1) v = W1[(size_t)k2 * N + n]; }
  dst[(size_t)f * 512 + threadIdx.x] = (_Float16)v;
}

__global__ void zero_kernel(float* __restrict__ p, int n) {
  int i = blockIdx.x * blockDim.x + threadIdx.x;
  if (i < n) p[i] = 0.f;
}

// ---------------- edge kernel ------------------------------------------------------------------
struct __align__(16) EdgeLds {
  _Float16 planeA[16 * 288];        // xs(0..159) | dotv(160..287); later overlay D2 + mv
  _Float16 planeXV[3][16 * 128];    // raw xv components (snd|rcv)
  _Float16 planeMS[16 * 64];        // silu feature scalars
  _Float16 planeG [16 * 64];        // sigmoid gates
  float asv[16];
  float avv[16][3];
  float valid[16];
  int   rcv[16];
  int   snd[16];
};

__global__ __launch_bounds__(EW * 32) void edge_kernel(
    const float* __restrict__ node_s, const float* __restrict__ node_v,
    const float* __restrict__ e_as, const float* __restrict__ e_av,
    const float* __restrict__ add_feat,
    const float* __restrict__ m1b, const float* __restrict__ m2b,
    const int* __restrict__ senders, const int* __restrict__ receivers,
    const _Float16* __restrict__ pk,
    float* __restrict__ out, int nE, int ntiles)
{
  __shared__ EdgeLds lds[EW];
  const int wave = threadIdx.x >> 5;
  const int lane = threadIdx.x & 31;
  const int tile = blockIdx.x * EW + wave;
  if (tile >= ntiles) return;
  EdgeLds& W = lds[wave];
  const int e0 = tile * 16;

  // per-row metadata
  if (lane < 16) {
    int e = e0 + lane;
    bool ok = (e < nE);
    int ec = ok ? e : (nE - 1);
    W.snd[lane] = senders[ec];
    W.rcv[lane] = receivers[ec];
    W.asv[lane] = ok ? e_as[ec] : 0.f;
    W.avv[lane][0] = ok ? e_av[ec * 3 + 0] : 0.f;
    W.avv[lane][1] = ok ? e_av[ec * 3 + 1] : 0.f;
    W.avv[lane][2] = ok ? e_av[ec * 3 + 2] : 0.f;
    W.valid[lane] = ok ? 1.f : 0.f;
  }
  wave_sync();

  const int row  = lane & 15;
  const int part = lane >> 4;      // 0 = sender half, 1 = receiver half

  // stage xs = [node_s[snd] | node_s[rcv] | add_feat | 0-pad]  (cols 0..159)
  {
    const int e = min(e0 + row, nE - 1);
    const int nid = part ? W.rcv[row] : W.snd[row];
    const float4* src4 = (const float4*)(node_s + (size_t)nid * 64);
    _Float16* dst = W.planeA + row * 288 + part * 64;
#pragma unroll
    for (int c4 = 0; c4 < 16; ++c4) {
      float4 f = src4[c4];
      v4h h = { (_Float16)f.x, (_Float16)f.y, (_Float16)f.z, (_Float16)f.w };
      *(v4h*)(dst + c4 * 4) = h;
    }
    if (part == 0) {
      _Float16* d2 = W.planeA + row * 288 + 128;
      d2[0] = (_Float16)add_feat[(size_t)e * 2 + 0];
      d2[1] = (_Float16)add_feat[(size_t)e * 2 + 1];
#pragma unroll
      for (int c = 2; c < 32; ++c) d2[c] = (_Float16)0.f;
    }
  }
  // stage xv planes + dotv = dot(xv, av)/sqrt3  (planeA cols 160..287)
  {
    const int nid = part ? W.rcv[row] : W.snd[row];
    const float* src = node_v + (size_t)nid * 192;
    const float av0 = W.avv[row][0], av1 = W.avv[row][1], av2 = W.avv[row][2];
#pragma unroll 4
    for (int o = 0; o < 64; ++o) {
      float x = src[o * 3 + 0], y = src[o * 3 + 1], z = src[o * 3 + 2];
      int col = part * 64 + o;
      W.planeXV[0][row * 128 + col] = (_Float16)x;
      W.planeXV[1][row * 128 + col] = (_Float16)y;
      W.planeXV[2][row * 128 + col] = (_Float16)z;
      W.planeA[row * 288 + 160 + col] = (_Float16)((x * av0 + y * av1 + z * av2) * RSQRT3);
    }
  }
  wave_sync();

  const int colN = row;            // C layout: col = lane&15
  const int mb = part * 8;         // C layout: rows mb..mb+7

  // cache per-C-row metadata in registers (a_s is a per-row diagonal scale -> pulled
  // out of the GEMMs entirely: s = a_s*(xs@Wss) + dot@Wvs etc.)
  float as8[8], vld8[8];
  int rcv8[8];
#pragma unroll
  for (int r = 0; r < 8; ++r) {
    as8[r]  = W.asv[mb + r];
    vld8[r] = W.valid[mb + r];
    rcv8[r] = W.rcv[mb + r];
  }

  // ---- m1 scalar path: s1 = (a_s*(xs@Wss) + dot@Wvs)*inv + b; silu / sigmoid ----
  for (int nt = 0; nt < 8; ++nt) {
    v8f cs = {}, cd = {};
    cs = WMMA(frag_a(W.planeA, 288, 0,   lane), frag_b(pk + OFF_M1S, nt * 9 + 0, lane), cs);
    cd = WMMA(frag_a(W.planeA, 288, 160, lane), frag_b(pk + OFF_M1S, nt * 9 + 5, lane), cd);
    cs = WMMA(frag_a(W.planeA, 288, 32,  lane), frag_b(pk + OFF_M1S, nt * 9 + 1, lane), cs);
    cd = WMMA(frag_a(W.planeA, 288, 192, lane), frag_b(pk + OFF_M1S, nt * 9 + 6, lane), cd);
    cs = WMMA(frag_a(W.planeA, 288, 64,  lane), frag_b(pk + OFF_M1S, nt * 9 + 2, lane), cs);
    cd = WMMA(frag_a(W.planeA, 288, 224, lane), frag_b(pk + OFF_M1S, nt * 9 + 7, lane), cd);
    cs = WMMA(frag_a(W.planeA, 288, 96,  lane), frag_b(pk + OFF_M1S, nt * 9 + 3, lane), cs);
    cd = WMMA(frag_a(W.planeA, 288, 256, lane), frag_b(pk + OFF_M1S, nt * 9 + 8, lane), cd);
    cs = WMMA(frag_a(W.planeA, 288, 128, lane), frag_b(pk + OFF_M1S, nt * 9 + 4, lane), cs);
    const int col = nt * 16 + colN;
    const float bias = m1b[col];
#pragma unroll
    for (int r = 0; r < 8; ++r) {
      const int m = mb + r;
      float s = (as8[r] * cs[r] + cd[r]) * INV_M1 + bias;
      if (nt < 4) W.planeMS[m * 64 + col] = (_Float16)(s * sigmoidf_(s));
      else        W.planeG[m * 64 + (col - 64)] = (_Float16)sigmoidf_(s);
    }
  }
  wave_sync();

  // ---- m1 T = xs @ W1sv (unscaled) ----
  v8f T[4];
  for (int nt = 0; nt < 4; ++nt) {
    v8f c = {};
#pragma unroll
    for (int kt = 0; kt < 5; ++kt)
      c = WMMA(frag_a(W.planeA, 288, kt * 32, lane), frag_b(pk + OFF_M1SV, nt * 5 + kt, lane), c);
    T[nt] = c;
  }

  // overlay: planeA (4608 halves) reused -> D2[1024] + mv[3][1024]
  _Float16* const D2 = W.planeA;
  _Float16* const MV[3] = { W.planeA + 1024, W.planeA + 2048, W.planeA + 3072 };

  // ---- m1 vector path: mv_i = (T*av_i + a_s*(xv_i@Wvv))*inv * gate ----
  for (int i = 0; i < 3; ++i) {
    float av8[8];
#pragma unroll
    for (int r = 0; r < 8; ++r) av8[r] = W.avv[mb + r][i];
    for (int nt = 0; nt < 4; ++nt) {
      v8f c = {};
#pragma unroll
      for (int kt = 0; kt < 4; ++kt)
        c = WMMA(frag_a(W.planeXV[i], 128, kt * 32, lane),
                 frag_b(pk + OFF_M1VV, nt * 4 + kt, lane), c);
      const int col = nt * 16 + colN;
#pragma unroll
      for (int r = 0; r < 8; ++r) {
        const int m = mb + r;
        float val = (as8[r] * c[r] + T[nt][r] * av8[r]) * INV_M1;
        val *= (float)W.planeG[m * 64 + col];
        MV[i][m * 64 + col] = (_Float16)val;
      }
    }
  }
  wave_sync();

  // ---- dot2 = dot(mv, av)/sqrt3 ----
  {
    const int cb = part * 32;
    const float av0 = W.avv[row][0], av1 = W.avv[row][1], av2 = W.avv[row][2];
#pragma unroll
    for (int c2 = 0; c2 < 32; ++c2) {
      const int col = cb + c2;
      float d = ((float)MV[0][row * 64 + col] * av0 +
                 (float)MV[1][row * 64 + col] * av1 +
                 (float)MV[2][row * 64 + col] * av2) * RSQRT3;
      D2[row * 64 + col] = (_Float16)d;
    }
  }
  wave_sync();

  // ---- m2 scalar path + scatter-add of feature scalars ----
  for (int nt = 0; nt < 8; ++nt) {
    v8f cs = {}, cd = {};
    cs = WMMA(frag_a(W.planeMS, 64, 0, lane),  frag_b(pk + OFF_M2S, nt * 4 + 0, lane), cs);
    cd = WMMA(frag_a(D2, 64, 0, lane),         frag_b(pk + OFF_M2S, nt * 4 + 2, lane), cd);
    cs = WMMA(frag_a(W.planeMS, 64, 32, lane), frag_b(pk + OFF_M2S, nt * 4 + 1, lane), cs);
    cd = WMMA(frag_a(D2, 64, 32, lane),        frag_b(pk + OFF_M2S, nt * 4 + 3, lane), cd);
    const int col = nt * 16 + colN;
    const float bias = m2b[col];
#pragma unroll
    for (int r = 0; r < 8; ++r) {
      const int m = mb + r;
      float s = (as8[r] * cs[r] + cd[r]) * INV_M2 + bias;
      if (nt < 4) {
        float v = s * sigmoidf_(s) * vld8[r];
        atomicAdd(out + (size_t)rcv8[r] * 256 + col, v);
      } else {
        W.planeG[m * 64 + (col - 64)] = (_Float16)sigmoidf_(s);
      }
    }
  }
  wave_sync();

  // ---- m2 T2 = ms @ W2sv ----
  v8f T2[4];
  for (int nt = 0; nt < 4; ++nt) {
    v8f c = {};
    c = WMMA(frag_a(W.planeMS, 64, 0, lane),  frag_b(pk + OFF_M2SV, nt * 2 + 0, lane), c);
    c = WMMA(frag_a(W.planeMS, 64, 32, lane), frag_b(pk + OFF_M2SV, nt * 2 + 1, lane), c);
    T2[nt] = c;
  }

  // ---- m2 vector path + scatter-add ----
  for (int i = 0; i < 3; ++i) {
    float av8[8];
#pragma unroll
    for (int r = 0; r < 8; ++r) av8[r] = W.avv[mb + r][i];
    for (int nt = 0; nt < 4; ++nt) {
      v8f c = {};
      c = WMMA(frag_a(MV[i], 64, 0, lane),  frag_b(pk + OFF_M2VV, nt * 2 + 0, lane), c);
      c = WMMA(frag_a(MV[i], 64, 32, lane), frag_b(pk + OFF_M2VV, nt * 2 + 1, lane), c);
      const int col = nt * 16 + colN;
#pragma unroll
      for (int r = 0; r < 8; ++r) {
        const int m = mb + r;
        float val = (as8[r] * c[r] + T2[nt][r] * av8[r]) * INV_M2;
        val *= (float)W.planeG[m * 64 + col] * vld8[r];
        atomicAdd(out + (size_t)rcv8[r] * 256 + 64 + col * 3 + i, val);
      }
    }
  }
}

// ---------------- node kernel ------------------------------------------------------------------
struct __align__(16) NodeLds {
  _Float16 planeA[16 * 256];        // xs(0..127) | dot(128..255); later overlay D2 + hv
  _Float16 planeXV[3][16 * 128];
  _Float16 planeMS[16 * 64];
  _Float16 planeG [16 * 64];
  float asv[16];
  float avv[16][3];
  float valid[16];
};

__global__ __launch_bounds__(EW * 32) void node_kernel(
    const float* __restrict__ node_s, const float* __restrict__ node_v,
    const float* __restrict__ n_as, const float* __restrict__ n_av,
    const float* __restrict__ u0b, const float* __restrict__ u1b,
    const _Float16* __restrict__ pk,
    float* __restrict__ out, int nN, int ntiles)
{
  __shared__ NodeLds lds[EW];
  const int wave = threadIdx.x >> 5;
  const int lane = threadIdx.x & 31;
  const int tile = blockIdx.x * EW + wave;
  if (tile >= ntiles) return;
  NodeLds& W = lds[wave];
  const int n0 = tile * 16;

  if (lane < 16) {
    int n = n0 + lane;
    bool ok = (n < nN);
    int nc = ok ? n : (nN - 1);
    W.asv[lane] = ok ? n_as[nc] : 0.f;
    W.avv[lane][0] = ok ? n_av[nc * 3 + 0] : 0.f;
    W.avv[lane][1] = ok ? n_av[nc * 3 + 1] : 0.f;
    W.avv[lane][2] = ok ? n_av[nc * 3 + 2] : 0.f;
    W.valid[lane] = ok ? 1.f : 0.f;
  }
  wave_sync();

  const int row  = lane & 15;
  const int part = lane >> 4;      // 0 = node feats, 1 = aggregated feats (read from out)
  const int n = min(n0 + row, nN - 1);

  // stage xs = [node_s | agg_s]  (agg lives in out[:, 0:64])
  {
    const float4* src4 = part ? (const float4*)(out + (size_t)n * 256)
                              : (const float4*)(node_s + (size_t)n * 64);
    _Float16* dst = W.planeA + row * 256 + part * 64;
#pragma unroll
    for (int c4 = 0; c4 < 16; ++c4) {
      float4 f = src4[c4];
      v4h h = { (_Float16)f.x, (_Float16)f.y, (_Float16)f.z, (_Float16)f.w };
      *(v4h*)(dst + c4 * 4) = h;
    }
  }
  // stage xv planes (node_v | agg_v from out[:, 64:256]) + dot plane
  {
    const float av0 = W.avv[row][0], av1 = W.avv[row][1], av2 = W.avv[row][2];
#pragma unroll 4
    for (int o = 0; o < 64; ++o) {
      const float* s = part ? (out + (size_t)n * 256 + 64 + o * 3)
                            : (node_v + (size_t)n * 192 + o * 3);
      float x = s[0], y = s[1], z = s[2];
      int col = part * 64 + o;
      W.planeXV[0][row * 128 + col] = (_Float16)x;
      W.planeXV[1][row * 128 + col] = (_Float16)y;
      W.planeXV[2][row * 128 + col] = (_Float16)z;
      W.planeA[row * 256 + 128 + col] = (_Float16)((x * av0 + y * av1 + z * av2) * RSQRT3);
    }
  }
  wave_sync();

  const int colN = row;
  const int mb = part * 8;

  float as8[8], vld8[8];
#pragma unroll
  for (int r = 0; r < 8; ++r) {
    as8[r]  = W.asv[mb + r];
    vld8[r] = W.valid[mb + r];
  }

  // ---- u0 scalar path: s = (a_s*(xs@Wss) + dot@Wvs)*inv + b ----
  for (int nt = 0; nt < 8; ++nt) {
    v8f cs = {}, cd = {};
#pragma unroll
    for (int kt = 0; kt < 4; ++kt) {
      cs = WMMA(frag_a(W.planeA, 256, kt * 32, lane),
                frag_b(pk + OFF_U0S, nt * 8 + kt, lane), cs);
      cd = WMMA(frag_a(W.planeA, 256, 128 + kt * 32, lane),
                frag_b(pk + OFF_U0S, nt * 8 + 4 + kt, lane), cd);
    }
    const int col = nt * 16 + colN;
    const float bias = u0b[col];
#pragma unroll
    for (int r = 0; r < 8; ++r) {
      const int m = mb + r;
      float s = (as8[r] * cs[r] + cd[r]) * INV_U0 + bias;
      if (nt < 4) W.planeMS[m * 64 + col] = (_Float16)(s * sigmoidf_(s));
      else        W.planeG[m * 64 + (col - 64)] = (_Float16)sigmoidf_(s);
    }
  }
  wave_sync();

  // ---- u0 T0 = xs @ u0_Wsv ----
  v8f T0[4];
  for (int nt = 0; nt < 4; ++nt) {
    v8f c = {};
#pragma unroll
    for (int kt = 0; kt < 4; ++kt)
      c = WMMA(frag_a(W.planeA, 256, kt * 32, lane), frag_b(pk + OFF_U0SV, nt * 4 + kt, lane), c);
    T0[nt] = c;
  }

  _Float16* const D2 = W.planeA;                                // overlay
  _Float16* const MV[3] = { W.planeA + 1024, W.planeA + 2048, W.planeA + 3072 };

  // ---- u0 vector path (gated): hv_i = (T0*av_i + a_s*(xv_i@Wvv))*inv * gate ----
  for (int i = 0; i < 3; ++i) {
    float av8[8];
#pragma unroll
    for (int r = 0; r < 8; ++r) av8[r] = W.avv[mb + r][i];
    for (int nt = 0; nt < 4; ++nt) {
      v8f c = {};
#pragma unroll
      for (int kt = 0; kt < 4; ++kt)
        c = WMMA(frag_a(W.planeXV[i], 128, kt * 32, lane),
                 frag_b(pk + OFF_U0VV, nt * 4 + kt, lane), c);
      const int col = nt * 16 + colN;
#pragma unroll
      for (int r = 0; r < 8; ++r) {
        const int m = mb + r;
        float val = (as8[r] * c[r] + T0[nt][r] * av8[r]) * INV_U0;
        val *= (float)W.planeG[m * 64 + col];
        MV[i][m * 64 + col] = (_Float16)val;
      }
    }
  }
  wave_sync();

  // ---- dot3 = dot(hv, av)/sqrt3 ----
  {
    const int cb = part * 32;
    const float av0 = W.avv[row][0], av1 = W.avv[row][1], av2 = W.avv[row][2];
#pragma unroll
    for (int c2 = 0; c2 < 32; ++c2) {
      const int col = cb + c2;
      float d = ((float)MV[0][row * 64 + col] * av0 +
                 (float)MV[1][row * 64 + col] * av1 +
                 (float)MV[2][row * 64 + col] * av2) * RSQRT3;
      D2[row * 64 + col] = (_Float16)d;
    }
  }
  wave_sync();

  // ---- u1 scalar path -> out[:, 0:64] = node_s + us ----
  for (int nt = 0; nt < 4; ++nt) {
    v8f cs = {}, cd = {};
    cs = WMMA(frag_a(W.planeMS, 64, 0, lane),  frag_b(pk + OFF_U1S, nt * 4 + 0, lane), cs);
    cd = WMMA(frag_a(D2, 64, 0, lane),         frag_b(pk + OFF_U1S, nt * 4 + 2, lane), cd);
    cs = WMMA(frag_a(W.planeMS, 64, 32, lane), frag_b(pk + OFF_U1S, nt * 4 + 1, lane), cs);
    cd = WMMA(frag_a(D2, 64, 32, lane),        frag_b(pk + OFF_U1S, nt * 4 + 3, lane), cd);
    const int col = nt * 16 + colN;
    const float bias = u1b[col];
#pragma unroll
    for (int r = 0; r < 8; ++r) {
      const int m = mb + r;
      if (vld8[r] > 0.f) {
        const int nn = n0 + m;
        float s = (as8[r] * cs[r] + cd[r]) * INV_U1 + bias;
        out[(size_t)nn * 256 + col] = node_s[(size_t)nn * 64 + col] + s;
      }
    }
  }

  // ---- u1 T1 = hs @ u1_Wsv ----
  v8f T1[4];
  for (int nt = 0; nt < 4; ++nt) {
    v8f c = {};
    c = WMMA(frag_a(W.planeMS, 64, 0, lane),  frag_b(pk + OFF_U1SV, nt * 2 + 0, lane), c);
    c = WMMA(frag_a(W.planeMS, 64, 32, lane), frag_b(pk + OFF_U1SV, nt * 2 + 1, lane), c);
    T1[nt] = c;
  }

  // ---- u1 vector path -> out[:, 64:256] = node_v + uv ----
  for (int i = 0; i < 3; ++i) {
    float av8[8];
#pragma unroll
    for (int r = 0; r < 8; ++r) av8[r] = W.avv[mb + r][i];
    for (int nt = 0; nt < 4; ++nt) {
      v8f c = {};
      c = WMMA(frag_a(MV[i], 64, 0, lane),  frag_b(pk + OFF_U1VV, nt * 2 + 0, lane), c);
      c = WMMA(frag_a(MV[i], 64, 32, lane), frag_b(pk + OFF_U1VV, nt * 2 + 1, lane), c);
      const int col = nt * 16 + colN;
#pragma unroll
      for (int r = 0; r < 8; ++r) {
        const int m = mb + r;
        if (vld8[r] > 0.f) {
          const int nn = n0 + m;
          float val = (as8[r] * c[r] + T1[nt][r] * av8[r]) * INV_U1;
          out[(size_t)nn * 256 + 64 + col * 3 + i] =
              node_v[(size_t)nn * 192 + col * 3 + i] + val;
        }
      }
    }
  }
}

// ---------------- launch -----------------------------------------------------------------------
extern "C" void kernel_launch(void* const* d_in, const int* in_sizes, int n_in,
                              void* d_out, int out_size, void* d_ws, size_t ws_size,
                              hipStream_t stream) {
  (void)n_in; (void)out_size; (void)ws_size;
  const float* node_s  = (const float*)d_in[0];
  const float* node_v  = (const float*)d_in[1];
  const float* n_as    = (const float*)d_in[2];
  const float* n_av    = (const float*)d_in[3];
  const float* e_as    = (const float*)d_in[4];
  const float* e_av    = (const float*)d_in[5];
  const float* addf    = (const float*)d_in[6];
  const float* m1_Wss  = (const float*)d_in[7];
  const float* m1_Wvs  = (const float*)d_in[8];
  const float* m1_Wsv  = (const float*)d_in[9];
  const float* m1_Wvv  = (const float*)d_in[10];
  const float* m1_b    = (const float*)d_in[11];
  const float* m2_Wss  = (const float*)d_in[12];
  const float* m2_Wvs  = (const float*)d_in[13];
  const float* m2_Wsv  = (const float*)d_in[14];
  const float* m2_Wvv  = (const float*)d_in[15];
  const float* m2_b    = (const float*)d_in[16];
  const float* u0_Wss  = (const float*)d_in[17];
  const float* u0_Wvs  = (const float*)d_in[18];
  const float* u0_Wsv  = (const float*)d_in[19];
  const float* u0_Wvv  = (const float*)d_in[20];
  const float* u0_b    = (const float*)d_in[21];
  const float* u1_Wss  = (const float*)d_in[22];
  const float* u1_Wvs  = (const float*)d_in[23];
  const float* u1_Wsv  = (const float*)d_in[24];
  const float* u1_Wvv  = (const float*)d_in[25];
  const float* u1_b    = (const float*)d_in[26];
  const int*   senders   = (const int*)d_in[27];
  const int*   receivers = (const int*)d_in[28];
  float* out = (float*)d_out;
  _Float16* pk = (_Float16*)d_ws;

  const int N = in_sizes[0] / 64;
  const int E = in_sizes[27];

  // out doubles as the aggregation buffer; zero it first (every call -> deterministic)
  {
    int tot = N * 256;
    zero_kernel<<<(tot + 255) / 256, 256, 0, stream>>>(out, tot);
  }

  // pack all weights into f16 WMMA B-fragments
  pack_kernel<<<8 * 9, 512, 0, stream>>>(m1_Wss, 130, 160, m1_Wvs, 128, 128, 9, pk + OFF_M1S);
  pack_kernel<<<4 * 5, 512, 0, stream>>>(m1_Wsv, 130, 160, nullptr, 0,  64, 5, pk + OFF_M1SV);
  pack_kernel<<<4 * 4, 512, 0, stream>>>(m1_Wvv, 128, 128, nullptr, 0,  64, 4, pk + OFF_M1VV);
  pack_kernel<<<8 * 4, 512, 0, stream>>>(m2_Wss,  64,  64, m2_Wvs, 64, 128, 4, pk + OFF_M2S);
  pack_kernel<<<4 * 2, 512, 0, stream>>>(m2_Wsv,  64,  64, nullptr, 0,  64, 2, pk + OFF_M2SV);
  pack_kernel<<<4 * 2, 512, 0, stream>>>(m2_Wvv,  64,  64, nullptr, 0,  64, 2, pk + OFF_M2VV);
  pack_kernel<<<8 * 8, 512, 0, stream>>>(u0_Wss, 128, 128, u0_Wvs, 128, 128, 8, pk + OFF_U0S);
  pack_kernel<<<4 * 4, 512, 0, stream>>>(u0_Wsv, 128, 128, nullptr, 0,  64, 4, pk + OFF_U0SV);
  pack_kernel<<<4 * 4, 512, 0, stream>>>(u0_Wvv, 128, 128, nullptr, 0,  64, 4, pk + OFF_U0VV);
  pack_kernel<<<4 * 4, 512, 0, stream>>>(u1_Wss,  64,  64, u1_Wvs, 64,  64, 4, pk + OFF_U1S);
  pack_kernel<<<4 * 2, 512, 0, stream>>>(u1_Wsv,  64,  64, nullptr, 0,  64, 2, pk + OFF_U1SV);
  pack_kernel<<<4 * 2, 512, 0, stream>>>(u1_Wvv,  64,  64, nullptr, 0,  64, 2, pk + OFF_U1VV);

  // edge message kernel (scatter-adds into out)
  {
    int tiles = (E + 15) / 16;
    int blocks = (tiles + EW - 1) / EW;
    edge_kernel<<<blocks, EW * 32, 0, stream>>>(node_s, node_v, e_as, e_av, addf,
                                                m1_b, m2_b, senders, receivers, pk,
                                                out, E, tiles);
  }
  // node update kernel (reads agg from out, writes final result in place)
  {
    int tiles = (N + 15) / 16;
    int blocks = (tiles + EW - 1) / EW;
    node_kernel<<<blocks, EW * 32, 0, stream>>>(node_s, node_v, n_as, n_av,
                                                u0_b, u1_b, pk, out, N, tiles);
  }
}
